// MSCNModel_21655225106657
// MI455X (gfx1250) — compile-verified, matrix operand order checked by reading, not run on hardware
//
#include <hip/hip_runtime.h>

typedef __attribute__((ext_vector_type(16))) _Float16 v16h;
typedef __attribute__((ext_vector_type(8)))  _Float16 v8h;
typedef __attribute__((ext_vector_type(8)))  float    v8f;

#define P_NODES 20000
#define HDIM    128
#define CDIM    512           // 4*H concatenated
#define CHUNK_K 128           // K-chunk staged in LDS (4 slabs of 32)
#define SB_ELEMS (4 * 8 * 32 * 16)   // slabs * ntiles * lanes * halves = 16384 (32 KB)

// K index inside a 32-wide slab for half-element pair j (0..7), per CDNA5
// 16-bit A-matrix 16x32 layout: j<4 -> K = half*8 + 2j ; j>=4 -> 16 + half*8 + 2(j-4)
__device__ __forceinline__ int kmap(int j, int half) {
    return ((j >> 2) << 4) + (half << 3) + ((j & 3) << 1);
}

// Stage a K-chunk of W (fp32 [Kdim,128]) into LDS as f16, pre-swizzled into the
// exact WMMA B-fragment layout: sB[(s*8+nt)*32 + lane][16 halves].
__device__ __forceinline__ void stage_B(const float* __restrict__ W, int Kdim,
                                        int kc, int slabs, _Float16* sB)
{
    for (int fi = threadIdx.x; fi < slabs * 8 * 32; fi += blockDim.x) {
        const int lane_s = fi & 31;
        const int nt     = (fi >> 5) & 7;
        const int s      = fi >> 8;
        const int m      = lane_s & 15;
        const int half   = lane_s >> 4;
        const float* base = W + (size_t)(kc + s * 32) * HDIM + nt * 16 + m;
        v16h v;
#pragma unroll
        for (int j = 0; j < 8; ++j) {
            const int k = kmap(j, half);
            v[2 * j]     = (_Float16)base[(size_t)k * HDIM];
            v[2 * j + 1] = (_Float16)base[(size_t)(k + 1) * HDIM];
        }
        *(v16h*)&sB[(size_t)fi * 16] = v;   // 32B aligned -> ds_store_b128 x2
    }
}

// ---------------------------------------------------------------------------
// Encoder: enc = relu(feat @ W + b), fused segment-sum scatter + counts.
// One wave per 16-row block; full 128-col output (8 WMMA accumulators).
// ---------------------------------------------------------------------------
__global__ void encoder_pool_kernel(const float* __restrict__ feat,  // [N, K]
                                    const int*   __restrict__ idx,   // [N]
                                    const float* __restrict__ W,     // [K, 128]
                                    const float* __restrict__ bias,  // [128]
                                    float* __restrict__ pooled,      // [P, 512]
                                    float* __restrict__ counts,      // [4, P]
                                    int Nrows, int Kdim, int colOff, int typeSlot)
{
    __shared__ _Float16 sB[SB_ELEMS];

    const int lane   = threadIdx.x & 31;
    const int wave   = threadIdx.x >> 5;
    const int rb     = blockIdx.x * 8 + wave;        // 16-row block id (wave-uniform)
    const bool active = (rb * 16 < Nrows);
    const int row0   = rb * 16;
    const int m      = lane & 15;
    const int half   = lane >> 4;

    const v8f vzero = {0.f, 0.f, 0.f, 0.f, 0.f, 0.f, 0.f, 0.f};
    v8f acc[8];
#pragma unroll
    for (int nt = 0; nt < 8; ++nt) acc[nt] = vzero;

    for (int kc = 0; kc < Kdim; kc += CHUNK_K) {
        const int rem   = (Kdim - kc) / 32;
        const int slabs = rem < 4 ? rem : 4;
        __syncthreads();
        stage_B(W, Kdim, kc, slabs, sB);
        __syncthreads();

        if (active) {
            for (int s = 0; s < slabs; ++s) {
                // A slab: two contiguous 8-float runs per lane -> 4x b128
                const float* arow = feat + (size_t)(row0 + m) * Kdim + kc + s * 32 + half * 8;
                const float4 r0 = *(const float4*)(arow);
                const float4 r1 = *(const float4*)(arow + 4);
                const float4 r2 = *(const float4*)(arow + 16);
                const float4 r3 = *(const float4*)(arow + 20);
                v16h a;
                a[0]  = (_Float16)r0.x;  a[1]  = (_Float16)r0.y;
                a[2]  = (_Float16)r0.z;  a[3]  = (_Float16)r0.w;
                a[4]  = (_Float16)r1.x;  a[5]  = (_Float16)r1.y;
                a[6]  = (_Float16)r1.z;  a[7]  = (_Float16)r1.w;
                a[8]  = (_Float16)r2.x;  a[9]  = (_Float16)r2.y;
                a[10] = (_Float16)r2.z;  a[11] = (_Float16)r2.w;
                a[12] = (_Float16)r3.x;  a[13] = (_Float16)r3.y;
                a[14] = (_Float16)r3.z;  a[15] = (_Float16)r3.w;
#pragma unroll
                for (int nt = 0; nt < 8; ++nt) {
                    const v16h b = *(const v16h*)&sB[(size_t)((s * 8 + nt) * 32 + lane) * 16];
                    acc[nt] = __builtin_amdgcn_wmma_f32_16x16x32_f16(
                        false, a, false, b, (short)0, acc[nt], false, false);
                }
            }
        }
    }

    if (!active) return;

    // accumulator VGPR r holds row M = r + 8*half (C/D layout), col N = m + 16*nt
    int plan[8];
#pragma unroll
    for (int r = 0; r < 8; ++r) plan[r] = idx[row0 + half * 8 + r];

#pragma unroll
    for (int nt = 0; nt < 8; ++nt) {
        const float bval = bias[nt * 16 + m];
#pragma unroll
        for (int r = 0; r < 8; ++r) {
            float v = acc[nt][r] + bval;
            v = v > 0.f ? v : 0.f;
            atomicAdd(&pooled[(size_t)plan[r] * CDIM + colOff + nt * 16 + m], v);
        }
    }
    if (m == 0) {   // count each source row exactly once (half 0 -> rows 0..7, half 1 -> 8..15)
#pragma unroll
        for (int r = 0; r < 8; ++r)
            atomicAdd(&counts[(size_t)typeSlot * P_NODES + plan[r]], 1.0f);
    }
}

// ---------------------------------------------------------------------------
// MLP GEMM: out = relu(A_f16 @ W_f32 + b) stored f16.  A: [M, Kdim] row-major.
// ---------------------------------------------------------------------------
__global__ void gemm_relu_kernel(const _Float16* __restrict__ A,   // [M, Kdim]
                                 const float* __restrict__ W,      // [Kdim, 128]
                                 const float* __restrict__ bias,   // [128]
                                 _Float16* __restrict__ out,       // [M, 128]
                                 int Mrows, int Kdim)
{
    __shared__ _Float16 sB[SB_ELEMS];

    const int lane   = threadIdx.x & 31;
    const int wave   = threadIdx.x >> 5;
    const int rb     = blockIdx.x * 8 + wave;
    const bool active = (rb * 16 < Mrows);
    const int row0   = rb * 16;
    const int m      = lane & 15;
    const int half   = lane >> 4;

    const v8f vzero = {0.f, 0.f, 0.f, 0.f, 0.f, 0.f, 0.f, 0.f};
    v8f acc[8];
#pragma unroll
    for (int nt = 0; nt < 8; ++nt) acc[nt] = vzero;

    for (int kc = 0; kc < Kdim; kc += CHUNK_K) {
        const int rem   = (Kdim - kc) / 32;
        const int slabs = rem < 4 ? rem : 4;
        __syncthreads();
        stage_B(W, Kdim, kc, slabs, sB);
        __syncthreads();

        if (active) {
            for (int s = 0; s < slabs; ++s) {
                // A slab (f16): two contiguous 8-half runs per lane -> 2x b128
                const _Float16* arow = A + (size_t)(row0 + m) * Kdim + kc + s * 32 + half * 8;
                const v8h lo = *(const v8h*)(arow);
                const v8h hi = *(const v8h*)(arow + 16);
                v16h a;
#pragma unroll
                for (int i = 0; i < 8; ++i) { a[i] = lo[i]; a[8 + i] = hi[i]; }
#pragma unroll
                for (int nt = 0; nt < 8; ++nt) {
                    const v16h b = *(const v16h*)&sB[(size_t)((s * 8 + nt) * 32 + lane) * 16];
                    acc[nt] = __builtin_amdgcn_wmma_f32_16x16x32_f16(
                        false, a, false, b, (short)0, acc[nt], false, false);
                }
            }
        }
    }

    if (!active) return;

#pragma unroll
    for (int nt = 0; nt < 8; ++nt) {
        const float bval = bias[nt * 16 + m];
#pragma unroll
        for (int r = 0; r < 8; ++r) {
            float v = acc[nt][r] + bval;
            v = v > 0.f ? v : 0.f;
            out[(size_t)(row0 + half * 8 + r) * HDIM + nt * 16 + m] = (_Float16)v;
        }
    }
}

// ---------------------------------------------------------------------------
// BatchNorm statistics: one block per column, reduce over 20000 rows of
// (pooled / max(count,1)).  Biased variance, matching the reference.
// ---------------------------------------------------------------------------
__global__ void bn_stats_kernel(const float* __restrict__ pooled,
                                const float* __restrict__ counts,
                                float* __restrict__ mu, float* __restrict__ var)
{
    const int c   = blockIdx.x;      // 0..511
    const int seg = c >> 7;
    float s = 0.f, ss = 0.f;
    for (int p = threadIdx.x; p < P_NODES; p += blockDim.x) {
        const float cnt = counts[(size_t)seg * P_NODES + p];
        const float x   = pooled[(size_t)p * CDIM + c] / fmaxf(cnt, 1.0f);
        s  += x;
        ss += x * x;
    }
    __shared__ float sh_s[256];
    __shared__ float sh_q[256];
    sh_s[threadIdx.x] = s;
    sh_q[threadIdx.x] = ss;
    __syncthreads();
    for (int off = 128; off > 0; off >>= 1) {
        if ((int)threadIdx.x < off) {
            sh_s[threadIdx.x] += sh_s[threadIdx.x + off];
            sh_q[threadIdx.x] += sh_q[threadIdx.x + off];
        }
        __syncthreads();
    }
    if (threadIdx.x == 0) {
        const float mean = sh_s[0] * (1.0f / P_NODES);
        mu[c]  = mean;
        var[c] = fmaxf(sh_q[0] * (1.0f / P_NODES) - mean * mean, 0.0f);
    }
}

__global__ void bn_norm_kernel(const float* __restrict__ pooled,
                               const float* __restrict__ counts,
                               const float* __restrict__ mu,
                               const float* __restrict__ var,
                               const float* __restrict__ gamma,
                               const float* __restrict__ beta,
                               _Float16* __restrict__ xbn)   // [P, 512] f16
{
    const int i = blockIdx.x * blockDim.x + threadIdx.x;
    if (i >= P_NODES * CDIM) return;
    const int c   = i & (CDIM - 1);
    const int p   = i >> 9;
    const int seg = c >> 7;
    const float cnt = counts[(size_t)seg * P_NODES + p];
    const float x   = pooled[i] / fmaxf(cnt, 1.0f);
    const float xn  = (x - mu[c]) * rsqrtf(var[c] + 1e-5f) * gamma[c] + beta[c];
    xbn[i] = (_Float16)xn;
}

// Final head: out[p] = h2[p,:] . W3 + b3
__global__ void head_kernel(const _Float16* __restrict__ h2,  // [P, 128]
                            const float* __restrict__ W3,     // [128]
                            const float* __restrict__ b3,     // [1]
                            float* __restrict__ out)          // [P]
{
    __shared__ float w[HDIM];
    if (threadIdx.x < HDIM) w[threadIdx.x] = W3[threadIdx.x];
    __syncthreads();
    const int p = blockIdx.x * blockDim.x + threadIdx.x;
    if (p >= P_NODES) return;
    const _Float16* row = h2 + (size_t)p * HDIM;
    float s = 0.f;
#pragma unroll 8
    for (int k = 0; k < HDIM; ++k) s += (float)row[k] * w[k];
    out[p] = s + b3[0];
}

// ---------------------------------------------------------------------------
extern "C" void kernel_launch(void* const* d_in, const int* in_sizes, int n_in,
                              void* d_out, int out_size, void* d_ws, size_t ws_size,
                              hipStream_t stream)
{
    // setup_inputs() order: per type t in (table, pred, agg, join):
    //   feat_t, idx_t, W_t, b_t   -> indices 4t .. 4t+3
    // then bn_gamma(16) bn_beta(17) W1(18) b1(19) W2(20) b2(21) W3(22) b3(23) num_plan(24)
    const float* feat[4];
    const int*   idx[4];
    const float* Wt[4];
    const float* bt[4];
    int          Nt[4], Kt[4];
    for (int t = 0; t < 4; ++t) {
        feat[t] = (const float*)d_in[4 * t + 0];
        idx[t]  = (const int*)  d_in[4 * t + 1];
        Wt[t]   = (const float*)d_in[4 * t + 2];
        bt[t]   = (const float*)d_in[4 * t + 3];
        Nt[t]   = in_sizes[4 * t + 1];
        Kt[t]   = in_sizes[4 * t + 0] / Nt[t];
    }
    const float* bn_gamma = (const float*)d_in[16];
    const float* bn_beta  = (const float*)d_in[17];
    const float* W1 = (const float*)d_in[18];
    const float* b1 = (const float*)d_in[19];
    const float* W2 = (const float*)d_in[20];
    const float* b2 = (const float*)d_in[21];
    const float* W3 = (const float*)d_in[22];
    const float* b3 = (const float*)d_in[23];

    // Workspace layout (256B aligned)
    char* ws = (char*)d_ws;
    size_t off = 0;
    auto take = [&](size_t bytes) { char* p = ws + off; off = (off + bytes + 255) & ~(size_t)255; return p; };
    float*    pooled = (float*)   take((size_t)P_NODES * CDIM * sizeof(float)); // 40.96 MB
    float*    counts = (float*)   take((size_t)4 * P_NODES * sizeof(float));    // 320 KB
    float*    mu     = (float*)   take(CDIM * sizeof(float));
    float*    var    = (float*)   take(CDIM * sizeof(float));
    _Float16* xbn    = (_Float16*)take((size_t)P_NODES * CDIM * sizeof(_Float16));
    _Float16* h1     = (_Float16*)take((size_t)P_NODES * HDIM * sizeof(_Float16));
    _Float16* h2     = (_Float16*)take((size_t)P_NODES * HDIM * sizeof(_Float16));
    (void)ws_size; (void)n_in; (void)out_size;

    // Zero the accumulation buffers (pooled + counts are contiguous at offset 0)
    hipMemsetAsync(d_ws, 0,
                   (size_t)P_NODES * CDIM * sizeof(float) + (size_t)4 * P_NODES * sizeof(float),
                   stream);

    // Encoders with fused segment-sum pooling (col offsets: table 0, pred 128, agg 256, join 384)
    for (int t = 0; t < 4; ++t) {
        const int rowblocks = (Nt[t] + 15) / 16;
        const int blocks    = (rowblocks + 7) / 8;
        encoder_pool_kernel<<<blocks, 256, 0, stream>>>(
            feat[t], idx[t], Wt[t], bt[t], pooled, counts, Nt[t], Kt[t], t * HDIM, t);
    }

    // BatchNorm (training-mode batch stats) -> f16 activations
    bn_stats_kernel<<<CDIM, 256, 0, stream>>>(pooled, counts, mu, var);
    bn_norm_kernel<<<(P_NODES * CDIM + 255) / 256, 256, 0, stream>>>(
        pooled, counts, mu, var, bn_gamma, bn_beta, xbn);

    // MLP head
    {
        const int rowblocks = P_NODES / 16;                       // 1250
        const int blocks    = (rowblocks + 7) / 8;
        gemm_relu_kernel<<<blocks, 256, 0, stream>>>(xbn, W1, b1, h1, P_NODES, CDIM);
        gemm_relu_kernel<<<blocks, 256, 0, stream>>>(h1,  W2, b2, h2, P_NODES, HDIM);
    }
    head_kernel<<<(P_NODES + 255) / 256, 256, 0, stream>>>(h2, W3, b3, (float*)d_out);
}
